// RNN_23922967839023
// MI455X (gfx1250) — matrix operand order repaired and checked by reading, not compiled, not run on hardware
//
#include <hip/hip_runtime.h>

// LSTM: BATCH=64, T=2048, IN=128, HID=256, gates=4*HID=1024
#define TLEN  2048
#define NB    64
#define HID_  256
#define G4    1024

typedef __attribute__((ext_vector_type(16))) __bf16        bf16x16;
typedef __attribute__((ext_vector_type(8)))  float         v8f;
typedef __attribute__((ext_vector_type(4)))  unsigned int  u32x4;   // clang vec: OK for NT builtins

union BFrag {
    bf16x16 v;
    u32x4   q[2];
};
union Pack8 {
    u32x4  q;
    __bf16 h[8];
};

__device__ __forceinline__ float sigf(float x) {
    return 1.0f / (1.0f + __expf(-x));
}
__device__ __forceinline__ float tanh_fast(float x) {
    float e = __expf(-2.0f * x);
    return (1.0f - e) / (1.0f + e);
}
__device__ __forceinline__ v8f wmma_bf16(const BFrag& a, const BFrag& b, v8f c) {
    return __builtin_amdgcn_wmma_f32_16x16x32_bf16(
        false, a.v, false, b.v, (short)0, c, false, false);
}

// ---------------------------------------------------------------------------
// Big time-parallel GEMM: G[t] = x_t @ W_ih + bias for ALL t at once.
// Output stored in bf16 in WMMA C-fragment order:
//   G[((t*256 + mb*64 + n16) * 256) + lane*8 + r]
// so the recurrent kernel initializes each accumulator with ONE b128 load.
// G is a 268 MB write-once stream -> non-temporal stores (keep weights + y0
// resident in the 192 MB L2 instead).
// ---------------------------------------------------------------------------
template<bool XBF16, int KX>
__global__ __launch_bounds__(256, 2)
void gate_gemm_kernel(const void*  __restrict__ xin_v,   // [B][T][KX] f32 or [T][B][KX] bf16
                      const __bf16* __restrict__ wT_ih,  // [1024][KX] bf16 (transposed)
                      const float* __restrict__ bias,    // [1024] f32
                      __bf16* __restrict__ G)            // fragment-major, see above
{
    const int tid  = threadIdx.x;
    const int lane = tid & 31;
    const int wv   = tid >> 5;
    const int lrow = lane & 15;
    const int hi   = lane >> 4;
    const int koff  = hi * 8;
    const int kboff = hi * 8;

    const size_t gid      = (size_t)blockIdx.x * 8 + wv;   // 0..131071
    const int    colgroup = (int)(gid & 15);               // 16 groups of 4 n16
    const size_t rowblock = gid >> 4;                      // 0..8191
    const int    t        = (int)(rowblock >> 2);
    const int    mb       = (int)(rowblock & 3);
    const int    brow     = mb * 16 + lrow;                // batch row

    const float*  xin_f = (const float*)xin_v;
    const __bf16* xin_b = (const __bf16*)xin_v;

    v8f acc[4];
#pragma unroll
    for (int g = 0; g < 4; ++g)
#pragma unroll
        for (int r = 0; r < 8; ++r) acc[g][r] = 0.0f;

#pragma unroll
    for (int kb = 0; kb < KX / 32; ++kb) {
        BFrag a;
        const int k0 = kb * 32 + koff;
        if constexpr (XBF16) {
            const __bf16* base = xin_b + ((size_t)t * NB + brow) * KX;
            a.q[0] = *reinterpret_cast<const u32x4*>(base + k0);
            a.q[1] = *reinterpret_cast<const u32x4*>(base + k0 + 16);
        } else {
            const float* base = xin_f + ((size_t)brow * TLEN + t) * KX;
#pragma unroll
            for (int j = 0; j < 8; ++j) a.v[j]     = (__bf16)base[k0 + j];
#pragma unroll
            for (int j = 0; j < 8; ++j) a.v[j + 8] = (__bf16)base[k0 + 16 + j];
        }
#pragma unroll
        for (int g = 0; g < 4; ++g) {
            const int n16 = colgroup * 4 + g;
            BFrag b;
            const __bf16* bp = wT_ih + (size_t)(n16 * 16 + lrow) * KX + kb * 32 + kboff;
            b.q[0] = *reinterpret_cast<const u32x4*>(bp);
            b.q[1] = *reinterpret_cast<const u32x4*>(bp + 16);
            acc[g] = wmma_bf16(a, b, acc[g]);
        }
    }

    // epilogue: fold bias, pack bf16, ONE coalesced non-temporal b128 store
#pragma unroll
    for (int g = 0; g < 4; ++g) {
        const int   n16 = colgroup * 4 + g;
        const float bn  = bias[(n16 << 4) + lrow];
        Pack8 pk;
#pragma unroll
        for (int r = 0; r < 8; ++r) pk.h[r] = (__bf16)(acc[g][r] + bn);
        __builtin_nontemporal_store(
            pk.q,
            reinterpret_cast<u32x4*>(
                G + ((size_t)t * 256 + mb * 64 + n16) * 256 + lane * 8));
    }
}

// ---------------------------------------------------------------------------
// Serial recurrence: single persistent workgroup (32 wave32 waves), h in LDS,
// c in registers. Wave w: nq = w&15 (coupled f/i/o/g column set), mq = w>>4,
// row blocks {mq, mq+2}. Per k-block: 2 A-frags (LDS) + 4 B-frags (L2) feed
// 8 WMMAs. Opaque s_mov_b32 per step stops LICM hoist/spill of weight loads.
// G fragments stream from HBM: prefetch next step's 8 fragments
// (global_prefetch_b8) and consume them with non-temporal loads.
// ---------------------------------------------------------------------------
template<bool OUT_F32>
__global__ __launch_bounds__(1024, 1)
void lstm_recur_kernel(const __bf16* __restrict__ G,      // precomputed x-gates (+bias)
                       const __bf16* __restrict__ wT_hh,  // [1024][256] bf16 (transposed)
                       __bf16* __restrict__ y_bf16,       // layer0 out [T][B][H] bf16
                       float*  __restrict__ y_f32,        // layer1 out [T][B][H] f32
                       float*  __restrict__ hn_out,       // [64][256] f32
                       float*  __restrict__ cn_out)       // [64][256] f32
{
    __shared__ __align__(16) __bf16 hbuf[NB * HID_];      // 32 KB persistent h

    const int tid  = threadIdx.x;
    const int lane = tid & 31;
    const int wv   = tid >> 5;          // 0..31
    const int nq   = wv & 15;           // hidden-col block (coupled gate set)
    const int mq   = wv >> 4;           // 0..1 -> row blocks {mq, mq+2}
    const int lrow = lane & 15;
    const int hi   = lane >> 4;
    const int koff  = hi * 8;
    const int kboff = hi * 8;

    for (int i = tid; i < NB * HID_; i += 1024) hbuf[i] = (__bf16)0.0f;
    __syncthreads();

    float creg[2][8];
#pragma unroll
    for (int p = 0; p < 2; ++p)
#pragma unroll
        for (int r = 0; r < 8; ++r) creg[p][r] = 0.0f;

    for (int s = 0; s < TLEN; ++s) {
        const int tt = TLEN - 1 - s;    // reversed-time input index

        // opaque zero: defeats LICM hoist/spill of loop-invariant weight loads
        int opq;
        asm volatile("s_mov_b32 %0, 0" : "=s"(opq));
        const __bf16* whh = wT_hh + opq;

        // ---- prefetch next step's G fragments (HBM stream) ----
        if (tt > 0) {
#pragma unroll
            for (int p = 0; p < 2; ++p)
#pragma unroll
                for (int g = 0; g < 4; ++g)
                    __builtin_prefetch(
                        G + ((size_t)(tt - 1) * 256 + (mq + 2 * p) * 64 + (nq + 16 * g)) * 256
                          + lane * 8,
                        0, 1);
        }

        // ---- init accumulators from precomputed gates (one NT b128 each) ----
        v8f acc[2][4];
#pragma unroll
        for (int p = 0; p < 2; ++p) {
            const int mb = mq + 2 * p;
#pragma unroll
            for (int g = 0; g < 4; ++g) {
                Pack8 pk;
                pk.q = __builtin_nontemporal_load(
                    reinterpret_cast<const u32x4*>(
                        G + ((size_t)tt * 256 + mb * 64 + (nq + 16 * g)) * 256 + lane * 8));
#pragma unroll
                for (int r = 0; r < 8; ++r) acc[p][g][r] = (float)pk.h[r];
            }
        }

        // ---- recurrent contribution: h @ w_hh ----
#pragma unroll
        for (int kb = 0; kb < HID_ / 32; ++kb) {
            BFrag a[2];
            const int k0 = kb * 32 + koff;
#pragma unroll
            for (int p = 0; p < 2; ++p) {
                const __bf16* hb = &hbuf[((mq + 2 * p) * 16 + lrow) * HID_];
                a[p].q[0] = *reinterpret_cast<const u32x4*>(hb + k0);      // ds_load_b128
                a[p].q[1] = *reinterpret_cast<const u32x4*>(hb + k0 + 16);
            }
#pragma unroll
            for (int g = 0; g < 4; ++g) {
                BFrag b;
                const __bf16* bp =
                    whh + (size_t)((nq + 16 * g) * 16 + lrow) * HID_ + kb * 32 + kboff;
                b.q[0] = *reinterpret_cast<const u32x4*>(bp);              // L2-resident
                b.q[1] = *reinterpret_cast<const u32x4*>(bp + 16);
                acc[0][g] = wmma_bf16(a[0], b, acc[0][g]);                 // B reused 2x
                acc[1][g] = wmma_bf16(a[1], b, acc[1][g]);
            }
        }

        __syncthreads();   // all h reads done before anyone rewrites h

        // ---- fused LSTM cell update on the C-fragment ----
        const int ncol = (nq << 4) + lrow;
#pragma unroll
        for (int p = 0; p < 2; ++p) {
#pragma unroll
            for (int r = 0; r < 8; ++r) {
                const int m = (mq + 2 * p) * 16 + r + (hi << 3);   // batch row (C layout)
                float f = acc[p][0][r];
                float i = acc[p][1][r];
                float o = acc[p][2][r];
                float g = acc[p][3][r];
                float cn = sigf(f) * creg[p][r] + sigf(i) * tanh_fast(g);
                float hn = sigf(o) * tanh_fast(cn);
                creg[p][r] = cn;
                hbuf[m * HID_ + ncol] = (__bf16)hn;
                const size_t yidx = ((size_t)s * NB + m) * HID_ + ncol;
                if constexpr (OUT_F32) {
                    __builtin_nontemporal_store(hn, y_f32 + yidx);  // write-once stream
                } else {
                    y_bf16[yidx] = (__bf16)hn;                      // re-read by GEMM1: keep in L2
                }
                if (s == TLEN - 1) {
                    hn_out[m * HID_ + ncol] = hn;
                    cn_out[m * HID_ + ncol] = cn;
                }
            }
        }
        __syncthreads();   // new h visible before next step's WMMA reads
    }
}

// One-time prep: w[K][N] f32 -> wT[N][K] bf16 (K-contiguous B-fragments).
__global__ void transpose_bf16_kernel(const float* __restrict__ src,
                                      __bf16* __restrict__ dst,
                                      int K, int N)
{
    int idx = blockIdx.x * blockDim.x + threadIdx.x;
    if (idx >= K * N) return;
    int n = idx / K;
    int k = idx - n * K;
    dst[idx] = (__bf16)src[(size_t)k * N + n];
}

extern "C" void kernel_launch(void* const* d_in, const int* in_sizes, int n_in,
                              void* d_out, int out_size, void* d_ws, size_t ws_size,
                              hipStream_t stream)
{
    (void)in_sizes; (void)n_in; (void)out_size; (void)ws_size;

    const float* x     = (const float*)d_in[0];
    const float* w_ih0 = (const float*)d_in[1];
    const float* w_hh0 = (const float*)d_in[2];
    const float* b0    = (const float*)d_in[3];
    const float* w_ih1 = (const float*)d_in[4];
    const float* w_hh1 = (const float*)d_in[5];
    const float* b1    = (const float*)d_in[6];
    float* out = (float*)d_out;

    // workspace layout (bf16): transposed weights + y0 + fragment-major G
    __bf16* ws     = (__bf16*)d_ws;
    __bf16* wT_ih0 = ws;                                   // 1024*128
    __bf16* wT_hh0 = wT_ih0 + (size_t)G4 * 128;            // 1024*256
    __bf16* wT_ih1 = wT_hh0 + (size_t)G4 * HID_;           // 1024*256
    __bf16* wT_hh1 = wT_ih1 + (size_t)G4 * HID_;           // 1024*256
    __bf16* y0     = wT_hh1 + (size_t)G4 * HID_;           // T*64*256
    __bf16* Gbuf   = y0 + (size_t)TLEN * NB * HID_;        // T*64*1024 (reused by both layers)

    // d_out layout: y1 [T,B,H] f32, then hn [2,64,256], then cn [2,64,256]
    const size_t Y = (size_t)TLEN * NB * HID_;
    float* y1  = out;
    float* hn0 = out + Y;
    float* hn1 = out + Y + (size_t)NB * HID_;
    float* cn0 = out + Y + (size_t)2 * NB * HID_;
    float* cn1 = out + Y + (size_t)3 * NB * HID_;

    {   // weight prep (one-time, trivially cheap)
        int n;
        n = G4 * 128;  transpose_bf16_kernel<<<(n + 255) / 256, 256, 0, stream>>>(w_ih0, wT_ih0, 128,  G4);
        n = G4 * HID_; transpose_bf16_kernel<<<(n + 255) / 256, 256, 0, stream>>>(w_hh0, wT_hh0, HID_, G4);
        n = G4 * HID_; transpose_bf16_kernel<<<(n + 255) / 256, 256, 0, stream>>>(w_ih1, wT_ih1, HID_, G4);
        n = G4 * HID_; transpose_bf16_kernel<<<(n + 255) / 256, 256, 0, stream>>>(w_hh1, wT_hh1, HID_, G4);
    }

    const int gemm_blocks = (TLEN * NB / 16) * 16 / 8;     // 131072 waves / 8 per block

    // Layer 0: time-parallel x-gates, then serial recurrence
    gate_gemm_kernel<false, 128><<<gemm_blocks, 256, 0, stream>>>(x, wT_ih0, b0, Gbuf);
    lstm_recur_kernel<false><<<1, 1024, 0, stream>>>(Gbuf, wT_hh0, y0, nullptr, hn0, cn0);

    // Layer 1: time-parallel y0-gates, then serial recurrence -> fp32 d_out
    gate_gemm_kernel<true, 256><<<gemm_blocks, 256, 0, stream>>>(y0, wT_ih1, b1, Gbuf);
    lstm_recur_kernel<true><<<1, 1024, 0, stream>>>(Gbuf, wT_hh1, nullptr, y1, hn1, cn1);
}
